// MyModel_32461362823875
// MI455X (gfx1250) — compile-verified, matrix operand order checked by reading, not run on hardware
//
#include <hip/hip_runtime.h>
#include <math.h>

#define T_   144
#define N_   66
#define F_   22
#define D_   1452      // N_*F_
#define H_   1000
#define G4   4000      // 4*H_
#define NBLK 40
#define UPB  25        // H_/NBLK hidden units per block

typedef float v2f  __attribute__((ext_vector_type(2)));
typedef float v8f  __attribute__((ext_vector_type(8)));
typedef __bf16 v16bf __attribute__((ext_vector_type(16)));

// ---- workspace layout (float offsets) ----
#define SEQ_OFF   0
#define XG_OFF    (SEQ_OFF + T_*D_)        // 144x4000 precomputed enc input gates (+biases)
#define HENC_OFF  (XG_OFF + T_*G4)
#define CENC_OFF  (HENC_OFF + H_)
#define HPRED_OFF (CENC_OFF + H_)
#define HREC_OFF  (HPRED_OFF + H_)
#define XPRED_OFF (HREC_OFF + H_)
#define XREC_OFF  (XPRED_OFF + D_)
#define LOSS_OFF  (XREC_OFF + D_)          // [0]=predict sum, [1]=recon sum
#define BAR_OFF   (LOSS_OFF + 2)           // 8 ints of barrier state

#ifndef __has_builtin
#define __has_builtin(x) 0
#endif
#if __has_builtin(__builtin_amdgcn_wmma_f32_16x16x4_f32)
#define USE_WMMA_F32 1
#else
#define USE_WMMA_F32 0
#endif

__device__ __forceinline__ float sigm(float x) { return 1.0f / (1.0f + expf(-x)); }

// Device-scope two-phase grid barrier over nblk blocks (persistent-kernel sync).
__device__ __forceinline__ void grid_sync(int* cnt, int* gen, int nblk) {
  __syncthreads();
  if (threadIdx.x == 0) {
    int g = __hip_atomic_load(gen, __ATOMIC_ACQUIRE, __HIP_MEMORY_SCOPE_AGENT);
    int arrived = __hip_atomic_fetch_add(cnt, 1, __ATOMIC_ACQ_REL, __HIP_MEMORY_SCOPE_AGENT);
    if (arrived == nblk - 1) {
      __hip_atomic_store(cnt, 0, __ATOMIC_RELAXED, __HIP_MEMORY_SCOPE_AGENT);
      __hip_atomic_fetch_add(gen, 1, __ATOMIC_ACQ_REL, __HIP_MEMORY_SCOPE_AGENT);
    } else {
      while (__hip_atomic_load(gen, __ATOMIC_ACQUIRE, __HIP_MEMORY_SCOPE_AGENT) == g) {
        __builtin_amdgcn_s_sleep(2);
      }
    }
  }
  __syncthreads();
}

// ---------------- init: zero all cross-launch state ----------------
__global__ void init_kernel(float* ws) {
  int tid = threadIdx.x;
  if (tid < 8) ((int*)(ws + BAR_OFF))[tid] = 0;
  if (tid < 2) ws[LOSS_OFF + tid] = 0.0f;
  for (int i = tid; i < H_; i += 256) ws[HENC_OFF + i] = 0.0f;  // enc h0 = 0
}

// ---------------- GAT: one timestep per block ----------------
__global__ __launch_bounds__(256) void gat_kernel(
    const float* __restrict__ in, const int* __restrict__ adj,
    const float* __restrict__ gatW, const float* __restrict__ gata,
    float* __restrict__ seq) {
  __shared__ float in_s[N_ * F_];
  __shared__ float Wh_s[N_ * F_];
  __shared__ float W_s[F_ * F_];
  __shared__ float a1_s[F_], a2_s[F_];
  __shared__ float s1_s[N_], s2_s[N_];
  __shared__ float e_s[N_ * N_];
  int t = blockIdx.x, tid = threadIdx.x;
  const float* inrow = in + (size_t)t * D_;
  for (int i = tid; i < N_ * F_; i += 256) in_s[i] = inrow[i];
  for (int i = tid; i < F_ * F_; i += 256) W_s[i] = gatW[i];
  if (tid < F_) { a1_s[tid] = gata[tid]; a2_s[tid] = gata[F_ + tid]; }
  __syncthreads();
  for (int i = tid; i < N_ * F_; i += 256) {
    int n = i / F_, g = i % F_;
    float acc = 0.0f;
    for (int f = 0; f < F_; f++) acc += in_s[n * F_ + f] * W_s[f * F_ + g];
    Wh_s[i] = acc;
  }
  __syncthreads();
  for (int n = tid; n < N_; n += 256) {
    float a1 = 0.0f, a2 = 0.0f;
    for (int g = 0; g < F_; g++) { float w = Wh_s[n * F_ + g]; a1 += w * a1_s[g]; a2 += w * a2_s[g]; }
    s1_s[n] = a1; s2_s[n] = a2;
  }
  __syncthreads();
  for (int i = tid; i < N_ * N_; i += 256) {
    int r = i / N_, c = i % N_;
    float v = s1_s[r] + s2_s[c];
    v = v > 0.0f ? v : 0.1f * v;                       // leaky_relu(0.1)
    e_s[i] = (adj[i] > 0) ? v : -9.0e15f;              // NEG mask
  }
  __syncthreads();
  for (int r = tid; r < N_; r += 256) {                // row softmax
    float m = -INFINITY;
    for (int c = 0; c < N_; c++) m = fmaxf(m, e_s[r * N_ + c]);
    float s = 0.0f;
    for (int c = 0; c < N_; c++) { float v = expf(e_s[r * N_ + c] - m); e_s[r * N_ + c] = v; s += v; }
    float inv = 1.0f / s;
    for (int c = 0; c < N_; c++) e_s[r * N_ + c] *= inv;
  }
  __syncthreads();
  float* outrow = seq + (size_t)t * D_;
  for (int i = tid; i < N_ * F_; i += 256) {
    int r = i / F_, f = i % F_;
    float acc = 0.0f;
    for (int j = 0; j < N_; j++) acc += e_s[r * N_ + j] * Wh_s[j * F_ + f];
    outrow[i] = acc > 0.0f ? acc : (expf(acc) - 1.0f); // elu
  }
}

// ---- WMMA GEMM: Xg = seq(144x1452) @ enc_Wih^T(1452x4000) + bih + bhh ----
// One 16x16 output tile per wave32. 144=9*16 M tiles, 4000=250*16 N tiles.
__global__ __launch_bounds__(128) void enc_gemm_wmma(
    const float* __restrict__ seq, const float* __restrict__ Wih,
    const float* __restrict__ bih, const float* __restrict__ bhh,
    float* __restrict__ Xg) {
  int tile = blockIdx.x * 4 + threadIdx.y;
  if (tile >= (T_ / 16) * (G4 / 16)) return;           // wave-uniform: EXEC stays all-ones
  int mT = tile / (G4 / 16), nT = tile % (G4 / 16);
  int lane = threadIdx.x;
  int mn = lane & 15;
  int m = mT * 16 + mn;
  int n = nT * 16 + mn;
  v8f c = {};
#if USE_WMMA_F32
  // f32 A 16x4 frag: lane<16 holds K=0,1 ; lane>=16 holds K=2,3 (M = lane&15). B mirrored.
  int khalf = (lane >> 4) << 1;
  const float* arow = seq + (size_t)m * D_ + khalf;
  const float* brow = Wih + (size_t)n * D_ + khalf;
  for (int k = 0; k < D_; k += 4) {
    v2f a = *(const v2f*)(arow + k);
    v2f b = *(const v2f*)(brow + k);
    __builtin_prefetch(brow + k + 64, 0, 0);           // global_prefetch_b8, 256B ahead
    c = __builtin_amdgcn_wmma_f32_16x16x4_f32(false, a, false, b, (short)0, c, false, false);
  }
#else
  // bf16 fallback (codegen-confirmed builtin). 16-bit A 16x32 layout:
  // lane<16: K = {k..k+7} U {k+16..k+23}; lane>=16: +8. B mirrored.
  const float* arow = seq + (size_t)m * D_;
  const float* brow = Wih + (size_t)n * D_;
  int kb = (lane >> 4) << 3;
  for (int k = 0; k < D_; k += 32) {
    v16bf a, b;
    for (int j = 0; j < 8; j++) {
      int k0 = k + kb + j, k1 = k + kb + 16 + j;
      a[j]     = (__bf16)(k0 < D_ ? arow[k0] : 0.0f);
      a[8 + j] = (__bf16)(k1 < D_ ? arow[k1] : 0.0f);
      b[j]     = (__bf16)(k0 < D_ ? brow[k0] : 0.0f);
      b[8 + j] = (__bf16)(k1 < D_ ? brow[k1] : 0.0f);
    }
    c = __builtin_amdgcn_wmma_f32_16x16x32_bf16(false, a, false, b, (short)0, c, false, false);
  }
#endif
  float bias = bih[n] + bhh[n];
  int mbase = mT * 16 + ((lane >> 4) ? 8 : 0);         // C/D layout: lanes>=16 hold M=8..15
#pragma unroll
  for (int v = 0; v < 8; v++) Xg[(size_t)(mbase + v) * G4 + nT * 16 + mn] = c[v] + bias;
}

// ---------------- persistent encoder LSTM: 40 blocks, 144 steps ----------------
__global__ __launch_bounds__(256) void enc_lstm(const float* __restrict__ Whh, float* ws) {
  __shared__ float h_s[H_];
  __shared__ float c_s[UPB];
  float* hglob = ws + HENC_OFF;
  const float* Xg = ws + XG_OFF;
  int* bar = (int*)(ws + BAR_OFF);
  int tid = threadIdx.x, lane = tid & 31, wid = tid >> 5;
  int u0 = blockIdx.x * UPB;
  if (tid < UPB) c_s[tid] = 0.0f;
  for (int t = 0; t < T_; t++) {
    for (int i = tid; i < H_; i += 256) h_s[i] = hglob[i];
    __syncthreads();
    for (int ul = wid; ul < UPB; ul += 8) {            // wave owns hidden unit -> 4 gate rows
      int u = u0 + ul;
      float dd[4];
      for (int g = 0; g < 4; g++) {
        const float* wr = Whh + (size_t)(g * H_ + u) * H_;
        float acc = 0.0f;
        for (int k = lane; k < H_; k += 32) acc += h_s[k] * wr[k];
        for (int off = 16; off > 0; off >>= 1) acc += __shfl_down(acc, off, 32);
        dd[g] = acc;
      }
      if (lane == 0) {
        const float* xr = Xg + (size_t)t * G4;
        float gi = sigm(dd[0] + xr[u]);
        float gf = sigm(dd[1] + xr[H_ + u]);
        float gg = tanhf(dd[2] + xr[2 * H_ + u]);
        float go = sigm(dd[3] + xr[3 * H_ + u]);
        float c2 = gf * c_s[ul] + gi * gg;
        c_s[ul] = c2;
        hglob[u] = go * tanhf(c2);
      }
    }
    grid_sync(bar + 0, bar + 1, NBLK);
  }
  if (tid < UPB) ws[CENC_OFF + u0 + tid] = c_s[tid];   // publish c_enc
}

// ---- persistent decoders: blocks [0,40)=predict, [40,80)=reconstruct, concurrent ----
__global__ __launch_bounds__(256) void dec_lstm(
    const float* pWih, const float* pWhh, const float* pbih, const float* pbhh,
    const float* pfcW, const float* pfcb,
    const float* rWih, const float* rWhh, const float* rbih, const float* rbhh,
    const float* rfcW, const float* rfcb,
    const float* __restrict__ target, const float* __restrict__ input, float* ws) {
  int dec = (blockIdx.x >= NBLK) ? 1 : 0;
  int b = blockIdx.x - dec * NBLK;
  const float* Wih = dec ? rWih : pWih;
  const float* Whh = dec ? rWhh : pWhh;
  const float* bih = dec ? rbih : pbih;
  const float* bhh = dec ? rbhh : pbhh;
  const float* fcW = dec ? rfcW : pfcW;
  const float* fcb = dec ? rfcb : pfcb;
  float* hbuf = ws + (dec ? HREC_OFF : HPRED_OFF);
  float* xbuf = ws + (dec ? XREC_OFF : XPRED_OFF);
  int* bar = (int*)(ws + BAR_OFF) + 2 + dec * 2;
  __shared__ float h_s[H_];
  __shared__ float x_s[D_];
  __shared__ float c_s[UPB];
  int tid = threadIdx.x, lane = tid & 31, wid = tid >> 5;
  int u0 = b * UPB;
  if (tid < UPB) c_s[tid] = ws[CENC_OFF + u0 + tid];
  for (int i = tid; i < H_; i += 256) h_s[i] = ws[HENC_OFF + i];
  __syncthreads();
  float lossAcc = 0.0f;
  for (int t = 0; t < T_; t++) {
    if (t > 0) {
      for (int i = tid; i < D_; i += 256) x_s[i] = xbuf[i];
      __syncthreads();
    }
    // phase 1: LSTM cell (x_0 == 0 -> skip x dot at t==0)
    for (int ul = wid; ul < UPB; ul += 8) {
      int u = u0 + ul;
      float dd[4];
      for (int g = 0; g < 4; g++) {
        int row = g * H_ + u;
        const float* wih = Wih + (size_t)row * D_;
        const float* whh = Whh + (size_t)row * H_;
        float acc = 0.0f;
        if (t > 0) for (int k = lane; k < D_; k += 32) acc += x_s[k] * wih[k];
        for (int k = lane; k < H_; k += 32) acc += h_s[k] * whh[k];
        for (int off = 16; off > 0; off >>= 1) acc += __shfl_down(acc, off, 32);
        dd[g] = acc;
      }
      if (lane == 0) {
        float gi = sigm(dd[0] + bih[u] + bhh[u]);
        float gf = sigm(dd[1] + bih[H_ + u] + bhh[H_ + u]);
        float gg = tanhf(dd[2] + bih[2 * H_ + u] + bhh[2 * H_ + u]);
        float go = sigm(dd[3] + bih[3 * H_ + u] + bhh[3 * H_ + u]);
        float c2 = gf * c_s[ul] + gi * gg;
        c_s[ul] = c2;
        hbuf[u] = go * tanhf(c2);
      }
    }
    grid_sync(bar, bar + 1, NBLK);
    for (int i = tid; i < H_; i += 256) h_s[i] = hbuf[i];
    __syncthreads();
    // phase 2: pred = h2 @ fcW^T + fcb; feed back as x; accumulate MSE
    for (int r = wid;; r += 8) {
      int dIdx = b + NBLK * r;
      if (dIdx >= D_) break;
      const float* wr = fcW + (size_t)dIdx * H_;
      float acc = 0.0f;
      for (int k = lane; k < H_; k += 32) acc += h_s[k] * wr[k];
      for (int off = 16; off > 0; off >>= 1) acc += __shfl_down(acc, off, 32);
      if (lane == 0) {
        float p = acc + fcb[dIdx];
        xbuf[dIdx] = p;
        float tgt = dec ? input[(size_t)(T_ - 1 - t) * D_ + dIdx]
                        : target[(size_t)t * D_ + dIdx];
        float df = p - tgt;
        lossAcc += df * df;
      }
    }
    grid_sync(bar, bar + 1, NBLK);
  }
  for (int off = 16; off > 0; off >>= 1) lossAcc += __shfl_down(lossAcc, off, 32);
  if (lane == 0) atomicAdd(ws + LOSS_OFF + dec, lossAcc);
}

__global__ void finalize_kernel(const float* __restrict__ ws, float* __restrict__ out) {
  if (threadIdx.x == 0) {
    float denom = (float)T_ * (float)D_;
    out[0] = ws[LOSS_OFF + 1] / denom;  // recon_loss
    out[1] = ws[LOSS_OFF + 0] / denom;  // predict_loss
  }
}

extern "C" void kernel_launch(void* const* d_in, const int* in_sizes, int n_in,
                              void* d_out, int out_size, void* d_ws, size_t ws_size,
                              hipStream_t stream) {
  const float* input  = (const float*)d_in[0];
  const int*   adj    = (const int*)d_in[1];
  const float* target = (const float*)d_in[2];
  const float* gatW   = (const float*)d_in[3];
  const float* gata   = (const float*)d_in[4];
  const float* eWih   = (const float*)d_in[5];
  const float* eWhh   = (const float*)d_in[6];
  const float* ebih   = (const float*)d_in[7];
  const float* ebhh   = (const float*)d_in[8];
  const float* pWih   = (const float*)d_in[9];
  const float* pWhh   = (const float*)d_in[10];
  const float* pbih   = (const float*)d_in[11];
  const float* pbhh   = (const float*)d_in[12];
  const float* pfcW   = (const float*)d_in[13];
  const float* pfcb   = (const float*)d_in[14];
  const float* rWih   = (const float*)d_in[15];
  const float* rWhh   = (const float*)d_in[16];
  const float* rbih   = (const float*)d_in[17];
  const float* rbhh   = (const float*)d_in[18];
  const float* rfcW   = (const float*)d_in[19];
  const float* rfcb   = (const float*)d_in[20];
  float* ws  = (float*)d_ws;
  float* out = (float*)d_out;

  init_kernel<<<1, 256, 0, stream>>>(ws);
  gat_kernel<<<T_, 256, 0, stream>>>(input, adj, gatW, gata, ws + SEQ_OFF);
  int tiles = (T_ / 16) * (G4 / 16);                     // 9 * 250 = 2250 wave-tiles
  enc_gemm_wmma<<<(tiles + 3) / 4, dim3(32, 4), 0, stream>>>(ws + SEQ_OFF, eWih, ebih, ebhh, ws + XG_OFF);
  enc_lstm<<<NBLK, 256, 0, stream>>>(eWhh, ws);
  dec_lstm<<<2 * NBLK, 256, 0, stream>>>(pWih, pWhh, pbih, pbhh, pfcW, pfcb,
                                         rWih, rWhh, rbih, rbhh, rfcW, rfcb,
                                         target, input, ws);
  finalize_kernel<<<1, 32, 0, stream>>>(ws, out);
}
